// NGCF_52785148068369
// MI455X (gfx1250) — matrix operand order repaired and checked by reading, not compiled, not run on hardware
//
#include <hip/hip_runtime.h>
#include <hip/hip_bf16.h>

typedef __attribute__((ext_vector_type(16))) _Float16 v16h;
typedef __attribute__((ext_vector_type(8)))  float    v8f;

#define USER_NUM 50000
#define ITEM_NUM 20000
#define NNODES   70000
#define DDIM     64
#define NEDGE    2000000
#define BATCH    16384

// ---------------------------------------------------------------- helpers ---

__device__ __forceinline__ v8f wmma_f16(v16h a, v16h b, v8f c) {
    // D = A(16x32 f16) * B(32x16 f16) + C(16x16 f32)
    return __builtin_amdgcn_wmma_f32_16x16x32_f16(
        /*neg_a=*/false, a, /*neg_b=*/false, b,
        /*c_mod=*/(short)0, c, /*reuse_a=*/false, /*reuse_b=*/false);
}

// A-matrix element k-offset for slot i (ISA 7.12.2, 16-bit A 16x32):
// lanes 0-15: VGPR0..3 -> K=0..7, VGPR4..7 -> K=16..23 ; lanes 16-31: +8
__device__ __forceinline__ int a_koff(int i, int kh) {
    return (i < 8) ? (kh * 8 + i) : (8 + kh * 8 + i);   // == 16 + kh*8 + (i-8)
}

// B tile (32x16) from row-major f32 weight W[K][ldn], at (k0, n0).
// lanes 0-15 hold K=k0+0..7,16..23; lanes 16-31 hold K=k0+8..15,24..31 (same as A).
__device__ __forceinline__ v16h load_b_f32(const float* __restrict__ W,
                                           int ldn, int k0, int n0, int lane) {
    int n = lane & 15, kh = lane >> 4;
    v16h b;
#pragma unroll
    for (int i = 0; i < 16; ++i) {
        int k = k0 + a_koff(i, kh);
        b[i] = (_Float16)W[k * ldn + n0 + n];
    }
    return b;
}

// ----------------------------------------------------------------- kernels --

// features = concat(user_emb, item_emb); lx = lx2 = 0
__global__ void __launch_bounds__(256)
ngcf_init_kernel(const float* __restrict__ ue, const float* __restrict__ ie,
                 float* __restrict__ feat, float* __restrict__ lx,
                 float* __restrict__ lx2) {
    int idx = blockIdx.x * blockDim.x + threadIdx.x;
    if (idx >= NNODES * DDIM) return;
    feat[idx] = (idx < USER_NUM * DDIM) ? ue[idx] : ie[idx - USER_NUM * DDIM];
    lx[idx]  = 0.0f;
    lx2[idx] = 0.0f;
}

// lx[row]  += val * feat[col];  lx2[row] += val * feat[col]^2
__global__ void __launch_bounds__(256)
ngcf_spmm_kernel(const int* __restrict__ erow, const int* __restrict__ ecol,
                 const float* __restrict__ eval, const float* __restrict__ feat,
                 float* __restrict__ lx, float* __restrict__ lx2) {
    int e = blockIdx.x * blockDim.x + threadIdx.x;
    if (e >= NEDGE) return;
    int   r = erow[e], c = ecol[e];
    float v = eval[e];
    const float4* frow = (const float4*)(feat + (size_t)c * DDIM);
    float* lr  = lx  + (size_t)r * DDIM;
    float* l2r = lx2 + (size_t)r * DDIM;
#pragma unroll 4
    for (int q = 0; q < DDIM / 4; ++q) {
        float4 f = frow[q];
        float fa[4] = {f.x, f.y, f.z, f.w};
#pragma unroll
        for (int j = 0; j < 4; ++j) {
            int d = q * 4 + j;
            __hip_atomic_fetch_add(&lr[d],  v * fa[j],
                                   __ATOMIC_RELAXED, __HIP_MEMORY_SCOPE_AGENT);
            __hip_atomic_fetch_add(&l2r[d], v * fa[j] * fa[j],
                                   __ATOMIC_RELAXED, __HIP_MEMORY_SCOPE_AGENT);
        }
    }
}

// h = leaky_relu((lx+feat)@Wg1 + bg1 + lx2@Wg2 + bg2), one wave per 16 rows
__global__ void __launch_bounds__(256)
ngcf_gcn_kernel(const float* __restrict__ feat, const float* __restrict__ lx,
                const float* __restrict__ lx2, const float* __restrict__ Wg1,
                const float* __restrict__ bg1, const float* __restrict__ Wg2,
                const float* __restrict__ bg2, float* __restrict__ h) {
    int wave = (blockIdx.x * blockDim.x + threadIdx.x) >> 5;
    int lane = threadIdx.x & 31;
    if (wave >= NNODES / 16) return;               // wave-uniform exit
    int rowbase = wave * 16;
    int m = lane & 15, kh = lane >> 4, n = lane & 15;

    const float* fp  = feat + (size_t)(rowbase + m) * DDIM;
    const float* lp  = lx   + (size_t)(rowbase + m) * DDIM;
    const float* l2p = lx2  + (size_t)(rowbase + m) * DDIM;

    v16h a1lo, a1hi, a2lo, a2hi;                   // K chunks [0,32) and [32,64)
#pragma unroll
    for (int i = 0; i < 16; ++i) {
        int k = a_koff(i, kh);
        a1lo[i] = (_Float16)(fp[k]      + lp[k]);
        a1hi[i] = (_Float16)(fp[k + 32] + lp[k + 32]);
        a2lo[i] = (_Float16)l2p[k];
        a2hi[i] = (_Float16)l2p[k + 32];
    }

#pragma unroll
    for (int t = 0; t < 4; ++t) {                  // 4 column tiles of 16
        int n0 = t * 16;
        v8f acc = {};
        acc = wmma_f16(a1lo, load_b_f32(Wg1, DDIM, 0,  n0, lane), acc);
        acc = wmma_f16(a1hi, load_b_f32(Wg1, DDIM, 32, n0, lane), acc);
        acc = wmma_f16(a2lo, load_b_f32(Wg2, DDIM, 0,  n0, lane), acc);
        acc = wmma_f16(a2hi, load_b_f32(Wg2, DDIM, 32, n0, lane), acc);
        float bb = bg1[n0 + n] + bg2[n0 + n];
#pragma unroll
        for (int r = 0; r < 8; ++r) {              // C/D: row = r + kh*8
            float v = acc[r] + bb;
            v = (v > 0.0f) ? v : 0.01f * v;        // leaky_relu 0.01
            h[(size_t)(rowbase + r + kh * 8) * DDIM + n0 + n] = v;
        }
    }
}

// 3-layer MLP over gathered [B,256] embeddings; one wave per 16 batch rows
__global__ void __launch_bounds__(256)
ngcf_mlp_kernel(const float* __restrict__ feat, const float* __restrict__ h,
                const float* __restrict__ W1, const float* __restrict__ b1,
                const float* __restrict__ W2, const float* __restrict__ b2,
                const float* __restrict__ W3, const float* __restrict__ b3,
                const int* __restrict__ userID, const int* __restrict__ itemID,
                float* __restrict__ out) {
    __shared__ _Float16 lds_e1[8][16 * 64];        // layer-1 activations (f16)
    __shared__ float    lds_e2[8][16 * 32];        // layer-2 activations (f32)
    int wlocal  = threadIdx.x >> 5;
    int wave    = blockIdx.x * 8 + wlocal;         // 1024 tiles / 8 = 128 blocks
    int lane    = threadIdx.x & 31;
    int rowbase = wave * 16;
    int m = lane & 15, kh = lane >> 4, n = lane & 15;

    int u  = userID[rowbase + m];
    int it = itemID[rowbase + m] + USER_NUM;
    const float* seg[4] = { feat + (size_t)u  * DDIM, h + (size_t)u  * DDIM,
                            feat + (size_t)it * DDIM, h + (size_t)it * DDIM };

    // ---- layer 1: [16,256] @ W1[256,64], K in 8 chunks of 32 ----
    v8f acc[4] = { v8f{}, v8f{}, v8f{}, v8f{} };
#pragma unroll
    for (int kk = 0; kk < 8; ++kk) {
        const float* sp = seg[kk >> 1];            // compile-time segment pick
        int obase = (kk & 1) * 32;
        v16h a;
#pragma unroll
        for (int i = 0; i < 16; ++i)
            a[i] = (_Float16)sp[obase + a_koff(i, kh)];
#pragma unroll
        for (int t = 0; t < 4; ++t)
            acc[t] = wmma_f16(a, load_b_f32(W1, DDIM, kk * 32, t * 16, lane), acc[t]);
    }
#pragma unroll
    for (int t = 0; t < 4; ++t) {
        float bb = b1[t * 16 + n];
#pragma unroll
        for (int r = 0; r < 8; ++r) {
            float v = acc[t][r] + bb;
            lds_e1[wlocal][(r + kh * 8) * 64 + t * 16 + n] =
                (_Float16)fmaxf(v, 0.0f);
        }
    }
    __syncthreads();

    // ---- layer 2: [16,64] @ W2[64,32] ----
    v8f acc2[2] = { v8f{}, v8f{} };
#pragma unroll
    for (int kc = 0; kc < 2; ++kc) {
        v16h a;
#pragma unroll
        for (int i = 0; i < 16; ++i)
            a[i] = lds_e1[wlocal][m * 64 + kc * 32 + a_koff(i, kh)];
#pragma unroll
        for (int t = 0; t < 2; ++t)
            acc2[t] = wmma_f16(a, load_b_f32(W2, 32, kc * 32, t * 16, lane), acc2[t]);
    }
#pragma unroll
    for (int t = 0; t < 2; ++t) {
        float bb = b2[t * 16 + n];
#pragma unroll
        for (int r = 0; r < 8; ++r) {
            float v = acc2[t][r] + bb;
            lds_e2[wlocal][(r + kh * 8) * 32 + t * 16 + n] = fmaxf(v, 0.0f);
        }
    }
    __syncthreads();

    // ---- layer 3: [16,32] @ W3[32,1] + b3 ----
    if (lane < 16) {
        float s = b3[0];
#pragma unroll
        for (int j = 0; j < 32; ++j)
            s += lds_e2[wlocal][lane * 32 + j] * W3[j];
        out[rowbase + lane] = s;
    }
}

// ------------------------------------------------------------------ launch --

extern "C" void kernel_launch(void* const* d_in, const int* in_sizes, int n_in,
                              void* d_out, int out_size, void* d_ws, size_t ws_size,
                              hipStream_t stream) {
    const float* user_emb = (const float*)d_in[0];
    const float* item_emb = (const float*)d_in[1];
    const int*   edge_row = (const int*)  d_in[2];
    const int*   edge_col = (const int*)  d_in[3];
    const float* edge_val = (const float*)d_in[4];
    const float* Wg1 = (const float*)d_in[5];
    const float* bg1 = (const float*)d_in[6];
    const float* Wg2 = (const float*)d_in[7];
    const float* bg2 = (const float*)d_in[8];
    const float* W1  = (const float*)d_in[9];
    const float* b1  = (const float*)d_in[10];
    const float* W2  = (const float*)d_in[11];
    const float* b2  = (const float*)d_in[12];
    const float* W3  = (const float*)d_in[13];
    const float* b3  = (const float*)d_in[14];
    const int* userID = (const int*)d_in[15];
    const int* itemID = (const int*)d_in[16];
    float* out = (float*)d_out;

    size_t nfeat = (size_t)NNODES * DDIM;          // 4.48M floats per buffer
    float* feat = (float*)d_ws;
    float* lx   = feat + nfeat;
    float* lx2  = lx   + nfeat;
    float* hbuf = lx2  + nfeat;

    int total = NNODES * DDIM;
    ngcf_init_kernel<<<(total + 255) / 256, 256, 0, stream>>>(
        user_emb, item_emb, feat, lx, lx2);
    ngcf_spmm_kernel<<<(NEDGE + 255) / 256, 256, 0, stream>>>(
        edge_row, edge_col, edge_val, feat, lx, lx2);
    int gtiles = NNODES / 16;                      // 4375 waves
    ngcf_gcn_kernel<<<(gtiles + 7) / 8, 256, 0, stream>>>(
        feat, lx, lx2, Wg1, bg1, Wg2, bg2, hbuf);
    ngcf_mlp_kernel<<<(BATCH / 16) / 8, 256, 0, stream>>>(
        feat, hbuf, W1, b1, W2, b2, W3, b3, userID, itemID, out);
}
